// LSTMLayer_87797721465193
// MI455X (gfx1250) — compile-verified, matrix operand order checked by reading, not run on hardware
//
#include <hip/hip_runtime.h>

// LSTM layer: B=128, T=512, D=512, H=1024, gates (i,f,g,o), f32 in/out.
// bf16 WMMA; x@W fused into the recurrent step; W/U pre-repacked into
// B-fragment order; per-step launch = grid barrier.
// Step kernel: block = 1 htile, 16 waves = 8 btiles x 2 K-halves; B operands
// staged via async global->LDS (double-buffered), A double-buffered in regs,
// all 4 gate B-fragments preloaded so the 4 WMMAs issue back-to-back,
// K-split partials reduced through LDS.

#define Bm  128
#define Tt  512
#define Dd  512
#define Hh  1024
#define G4H 4096

typedef __attribute__((ext_vector_type(16))) __bf16       v16bf;
typedef __attribute__((ext_vector_type(8)))  float        v8f;
typedef __attribute__((ext_vector_type(4)))  unsigned int v4u;
typedef __attribute__((ext_vector_type(4)))  int          v4i;

typedef v4i __attribute__((address_space(1)))* as1_v4i;
typedef v4i __attribute__((address_space(3)))* as3_v4i;

union Frag { v16bf v; v4u q[2]; };

#if defined(__HIP_DEVICE_COMPILE__) && \
    __has_builtin(__builtin_amdgcn_global_load_async_to_lds_b128) && \
    __has_builtin(__builtin_amdgcn_s_wait_asynccnt)
#define USE_ASYNC 1
#else
#define USE_ASYNC 0
#endif

__device__ __forceinline__ void async_cp16(const unsigned* gsrc, unsigned* ldst) {
#if USE_ASYNC
  __builtin_amdgcn_global_load_async_to_lds_b128((as1_v4i)gsrc, (as3_v4i)ldst, 0, 0);
#else
  *(v4u*)ldst = *(const v4u*)gsrc;
#endif
}

__device__ __forceinline__ void async_drain() {
#if USE_ASYNC
  __builtin_amdgcn_s_wait_asynccnt(0);
#endif
}

__device__ __forceinline__ unsigned bf16rne(float f) {
  unsigned u = __float_as_uint(f);
  return (u + 0x7FFFu + ((u >> 16) & 1u)) >> 16;   // round-to-nearest-even bf16
}

__device__ __forceinline__ float sigf(float x) {
  return __builtin_amdgcn_rcpf(1.0f + __expf(-x));
}
__device__ __forceinline__ float tanh_fast(float x) {
  return fmaf(2.0f, __builtin_amdgcn_rcpf(1.0f + __expf(-2.0f * x)), -1.0f);
}

// ---- one-time prep kernels -------------------------------------------------

__global__ void cvt_bf16_pk(const float* __restrict__ src,
                            unsigned* __restrict__ dst, long n2) {
  long i = (long)blockIdx.x * blockDim.x + threadIdx.x;
  if (i >= n2) return;
  float f0 = src[2 * i], f1 = src[2 * i + 1];
  dst[i] = bf16rne(f0) | (bf16rne(f1) << 16);
}

// Row-major [K,N] f32 -> bf16 WMMA-B fragments, fragment-contiguous:
// P[(ntile*KT + ktile)*256 + lane*8 + v]; lane n = lane&15,
// k = (v<4?0:16) + (lane>=16?8:0) + (v&3)*2 + p  (mirror of documented A layout)
__global__ void repack_B(const float* __restrict__ S, unsigned* __restrict__ P,
                         int K, int N, int total) {
  int idx = blockIdx.x * blockDim.x + threadIdx.x;
  if (idx >= total) return;
  int v    = idx & 7;
  int lane = (idx >> 3) & 31;
  int tile = idx >> 8;                 // = ntile*KT + ktile
  int KT   = K >> 5;
  int ktile = tile % KT;
  int ntile = tile / KT;
  int n = (ntile << 4) + (lane & 15);
  int k = (ktile << 5) + ((v < 4) ? 0 : 16) + ((lane >= 16) ? 8 : 0) + ((v & 3) << 1);
  float f0 = S[(size_t)k * N + n];
  float f1 = S[(size_t)(k + 1) * N + n];
  P[idx] = bf16rne(f0) | (bf16rne(f1) << 16);
}

__global__ void zero_u32(unsigned* __restrict__ p, int n) {
  int i = blockIdx.x * blockDim.x + threadIdx.x;
  if (i < n) p[i] = 0u;
}

// ---- recurrent step --------------------------------------------------------
// Grid: 64 blocks (one per htile) x 512 threads (16 waves).
// wave wv: btile = wv&7, half = wv>>3. 24 K-tiles per half:
//   half0: j<16 -> Wp ktile j (A from x); j>=16 -> Up ktile j-16 (A from h)
//   half1: Up ktile j+8 (A from h)
__global__ __launch_bounds__(512) void lstm_step(
    const unsigned short* __restrict__ xb,    // bf16 [B,T,D]
    const unsigned*       __restrict__ Wp,    // packed B-frags, KT=16
    const unsigned*       __restrict__ Up,    // packed B-frags, KT=32
    const float*          __restrict__ bias,  // f32 [4H]
    const unsigned short* __restrict__ hprev, // bf16 [B,H]
    unsigned short*       __restrict__ hnext, // bf16 [B,H]
    float*                __restrict__ c,     // f32 [B,H] (in-place)
    float*                __restrict__ out,   // f32 [B,T,H]
    int t) {
  __shared__ unsigned stage[2][2][1024];      // [half][buf][4 gates * 256 dw] 16KB
  __shared__ float    red[8][32][36];         // [btile][lane][gate*8+r], padded

  const int lane  = threadIdx.x & 31;
  const int wv    = threadIdx.x >> 5;         // 0..15
  const int btile = wv & 7;
  const int half  = wv >> 3;
  const int htile = blockIdx.x;               // 0..63
  const int m0    = btile << 4;
  const int lrow  = lane & 15;
  const int ksel  = (lane >> 4) << 3;
  const int n     = (htile << 4) + lrow;
  const bool producer = (btile == 0);         // waves 0 and 8 stage B

  const unsigned short* xrow = xb + ((size_t)(m0 + lrow) * Tt + t) * Dd + ksel;
  const unsigned short* hrow = hprev + (size_t)(m0 + lrow) * Hh + ksel;

  v8f acc[4];
#pragma unroll
  for (int g = 0; g < 4; ++g) {
    float bz = (half == 0) ? bias[g * Hh + n] : 0.0f;
#pragma unroll
    for (int r = 0; r < 8; ++r) acc[g][r] = bz;
  }

  auto stage_issue = [&](int j, int buf) {
    const unsigned* P; int KT, kt;
    if (half == 0) {
      if (j < 16) { P = Wp; KT = 16; kt = j; }
      else        { P = Up; KT = 32; kt = j - 16; }
    } else        { P = Up; KT = 32; kt = j + 8; }
    unsigned* dst = &stage[half][buf][0];
#pragma unroll
    for (int g = 0; g < 4; ++g) {
      const unsigned* src = P + ((size_t)(htile + g * 64) * KT + kt) * 256;
      async_cp16(src + lane * 4,       dst + g * 256 + lane * 4);
      async_cp16(src + 128 + lane * 4, dst + g * 256 + 128 + lane * 4);
    }
  };

  auto loadA = [&](int j) {
    const unsigned short* p;
    if (half == 0) p = (j < 16) ? (xrow + j * 32) : (hrow + (j - 16) * 32);
    else           p = hrow + (j + 8) * 32;
    Frag a;
    a.q[0] = *(const v4u*)p;
    a.q[1] = *(const v4u*)(p + 16);
    return a;
  };

  if (producer) { stage_issue(0, 0); async_drain(); }
  __syncthreads();
  Frag a_cur = loadA(0);
  int cur = 0;

  for (int j = 0; j < 24; ++j) {
    if (producer && (j + 1 < 24)) stage_issue(j + 1, cur ^ 1);  // prefetch B -> LDS
    Frag a_nxt = (j + 1 < 24) ? loadA(j + 1) : a_cur;           // prefetch A -> regs

    // load all 4 gate B-fragments first (one dscnt wait), then 4 WMMAs b2b
    Frag bw[4];
#pragma unroll
    for (int g = 0; g < 4; ++g) {
      const unsigned* sb = &stage[half][cur][g * 256 + lane * 8];
      bw[g].q[0] = *(const v4u*)sb;
      bw[g].q[1] = *(const v4u*)(sb + 4);
    }
#pragma unroll
    for (int g = 0; g < 4; ++g) {
      acc[g] = __builtin_amdgcn_wmma_f32_16x16x32_bf16(
          false, a_cur.v, false, bw[g].v, (short)0, acc[g], false, false);
    }

    if (producer) async_drain();   // next buffer's async copies landed
    __syncthreads();
    a_cur = a_nxt;
    cur ^= 1;
  }

  // ---- reduce K-split partials through LDS ----
  if (half == 1) {
#pragma unroll
    for (int g = 0; g < 4; ++g)
#pragma unroll
      for (int r = 0; r < 8; ++r) red[btile][lane][g * 8 + r] = acc[g][r];
  }
  __syncthreads();

  if (half == 0) {
#pragma unroll
    for (int g = 0; g < 4; ++g)
#pragma unroll
      for (int r = 0; r < 8; ++r) acc[g][r] += red[btile][lane][g * 8 + r];

    // ---- gates + state update (C/D layout: VGPR r -> M = r (+8 hi lanes)) ----
    const int mhi = (lane & 16) ? 8 : 0;
#pragma unroll
    for (int r = 0; r < 8; ++r) {
      int m = m0 + r + mhi;
      float iv = sigf(acc[0][r]);
      float fv = sigf(acc[1][r]);
      float gv = tanh_fast(acc[2][r]);
      float ov = sigf(acc[3][r]);
      size_t ci = (size_t)m * Hh + n;
      float cn = fmaf(fv, c[ci], iv * gv);
      c[ci] = cn;
      float hv = ov * tanh_fast(cn);
      hnext[ci] = (unsigned short)bf16rne(hv);
      out[((size_t)m * Tt + t) * Hh + n] = hv;
    }
  }
}

// ---- host ------------------------------------------------------------------
extern "C" void kernel_launch(void* const* d_in, const int* in_sizes, int n_in,
                              void* d_out, int out_size, void* d_ws, size_t ws_size,
                              hipStream_t stream) {
  (void)in_sizes; (void)n_in; (void)out_size; (void)ws_size;
  const float* x = (const float*)d_in[0];
  const float* W = (const float*)d_in[1];
  const float* U = (const float*)d_in[2];
  const float* b = (const float*)d_in[3];
  float* out = (float*)d_out;

  char* ws = (char*)d_ws;
  size_t o = 0;
  unsigned short* xb = (unsigned short*)(ws + o); o += (size_t)Bm * Tt * Dd * 2;  // 64 MiB
  unsigned*       Wp = (unsigned*)(ws + o);       o += (size_t)Dd * G4H * 2;      //  4 MiB
  unsigned*       Up = (unsigned*)(ws + o);       o += (size_t)Hh * G4H * 2;      //  8 MiB
  unsigned short* h0 = (unsigned short*)(ws + o); o += (size_t)Bm * Hh * 2;
  unsigned short* h1 = (unsigned short*)(ws + o); o += (size_t)Bm * Hh * 2;
  float*          c  = (float*)(ws + o);          o += (size_t)Bm * Hh * 4;

  long n2 = (long)Bm * Tt * Dd / 2;                   // 16,777,216 dwords
  cvt_bf16_pk<<<(unsigned)((n2 + 255) / 256), 256, 0, stream>>>(x, (unsigned*)xb, n2);

  int totW = (Dd / 32) * (G4H / 16) * 256;            // 1,048,576 dwords
  repack_B<<<(totW + 255) / 256, 256, 0, stream>>>(W, Wp, Dd, G4H, totW);
  int totU = (Hh / 32) * (G4H / 16) * 256;            // 2,097,152 dwords
  repack_B<<<(totU + 255) / 256, 256, 0, stream>>>(U, Up, Hh, G4H, totU);

  int zn = (int)(((size_t)Bm * Hh * 2 * 2 + (size_t)Bm * Hh * 4) / 4);  // h0,h1,c
  zero_u32<<<(zn + 255) / 256, 256, 0, stream>>>((unsigned*)h0, zn);

  for (int t = 0; t < Tt; ++t) {
    const unsigned short* hp = (t & 1) ? h1 : h0;
    unsigned short*       hn = (t & 1) ? h0 : h1;
    lstm_step<<<64, 512, 0, stream>>>(xb, Wp, Up, b, hp, hn, c, out, t);
  }
}
